// LinearReLU_19_18442589569842
// MI455X (gfx1250) — matrix operand order repaired and checked by reading, not compiled, not run on hardware
//
#include <hip/hip_runtime.h>
#include <stdint.h>

// ---------------------------------------------------------------------------
// Fused MLP forward for MI455X (gfx1250, wave32, WMMA + Tensor Data Mover).
// x [65536,784] f32 -> L0(784->124) -> relu L1(124->32) -> relu+b L3(32->22)
// -> relu+b L5(22->21) -> L7(21->10) -> log_softmax.
// Memory-bound on the 205MB x read (~9us at 23.3 TB/s); all matmuls use
// v_wmma_f32_16x16x32_bf16; x/B tiles are DMA'd into LDS by the TDM with
// double-buffered overlap.
// ---------------------------------------------------------------------------

typedef __attribute__((ext_vector_type(16))) __bf16 v16bf;
typedef __attribute__((ext_vector_type(8)))  float  v8f;

union Frag {
  uint32_t u[8];
  v16bf    v;
};

// Cheap bf16 conversions (round-half-up; NaN edge ignored).
static __device__ __forceinline__ uint16_t bf16_1(float f) {
  return (uint16_t)((__float_as_uint(f) + 0x8000u) >> 16);
}
// Pack two f32 -> bf16x2 with one v_perm_b32.
static __device__ __forceinline__ uint32_t pack_bf16(float lo, float hi) {
  uint32_t a = __float_as_uint(lo) + 0x8000u;
  uint32_t b = __float_as_uint(hi) + 0x8000u;
  return __builtin_amdgcn_perm(b, a, 0x07060302u);  // {b.b3,b.b2,a.b3,a.b2}
}
// RNE version for the (cold) weight-pack kernel.
static __device__ __forceinline__ uint16_t bf16_rne(float f) {
  uint32_t u = __float_as_uint(f);
  u += 0x7FFFu + ((u >> 16) & 1u);
  return (uint16_t)(u >> 16);
}

static __device__ __forceinline__ v8f wmma_bf16(const Frag& a, const Frag& b, v8f c) {
  return __builtin_amdgcn_wmma_f32_16x16x32_bf16(false, a.v, false, b.v,
                                                 (short)0, c, false, false);
}

// Workspace layout (dwords) for prepacked bf16 weight fragments.
#define OFF0 0        // L0: KT=25, NT=8  -> 51200 dw
#define OFF1 51200    // L1: KT=4,  NT=2  -> 2048  dw
#define OFF3 53248    // L3: 512 dw
#define OFF5 53760    // L5: 512 dw
#define OFF7 54272    // L7: 256 dw

// ---------------------------------------------------------------------------
// Tensor Data Mover helpers (CDNA5 TDM, D# per ISA 08_async_tensor §8.3/8.4).
// This toolchain exposes the 6-arg builtin:
//   (uint32x4 group0, int32x8 group1, int32x4 group2, int32x4 group3,
//    int32x8 <extra>, i32 cpol)
// Groups 2/3 and the extra vector are zero for 2D non-iterating descriptors.
// ---------------------------------------------------------------------------
#if __has_builtin(__builtin_amdgcn_tensor_load_to_lds)
#define HAVE_TDM 1
typedef __attribute__((ext_vector_type(4))) unsigned int v4u;
typedef __attribute__((ext_vector_type(8))) int v8i;
typedef __attribute__((ext_vector_type(4))) int v4i;

// 1D contiguous copy: ndw dwords global -> LDS.
static __device__ __forceinline__ void tdm_1d(uint32_t lds_addr, const void* g,
                                              uint32_t ndw) {
  uint64_t ga = (uint64_t)(uintptr_t)g;
  v4u g0 = {1u, lds_addr, (uint32_t)ga, (uint32_t)(ga >> 32) | (2u << 30)};
  // data_size=4B; tensor_dim0=ndw; tensor_dim1=1; tile_dim0=ndw; stride=ndw
  v8i g1 = {0x20000, (int)(ndw << 16), 0x10000, (int)(ndw << 16),
            0, (int)ndw, 0, 0};
  v4i z4 = {0, 0, 0, 0};
  v8i z8 = {0, 0, 0, 0, 0, 0, 0, 0};
  __builtin_amdgcn_tensor_load_to_lds(g0, g1, z4, z4, z8, 0);
}

// 2D x-tile: 16 rows x 32 f32, row stride 784 f32; LDS rows padded to 34
// dwords via pad_interval=32dw (code 4), pad_amount=2dw (code 1).
// rem = remaining valid columns from tile start (OOB columns read as zero).
static __device__ __forceinline__ void tdm_x(uint32_t lds_addr, const float* g,
                                             int rem) {
  uint64_t ga = (uint64_t)(uintptr_t)g;
  v4u g0 = {1u, lds_addr, (uint32_t)ga, (uint32_t)(ga >> 32) | (2u << 30)};
  // dword0: data_size=2(4B) | pad_enable | pad_interval=4 | pad_amount=1
  v8i g1 = {(int)0x03120000,
            (int)(((uint32_t)rem & 0xFFFFu) << 16),  // tensor_dim0[15:0]
            (int)(16u << 16),                        // tensor_dim1 = 16 rows
            (int)(32u << 16),                        // tile_dim0 = 32
            16,                                      // tile_dim1 = 16
            784,                                     // tensor_dim0_stride
            0, 0};
  v4i z4 = {0, 0, 0, 0};
  v8i z8 = {0, 0, 0, 0, 0, 0, 0, 0};
  __builtin_amdgcn_tensor_load_to_lds(g0, g1, z4, z4, z8, 0);
}
#else
#define HAVE_TDM 0
#endif

// ---------------------------------------------------------------------------
// Prep: pack w[N,K] f32 -> bf16 WMMA B-fragments, tile order [kt][nt],
// per-tile [lane][vgpr] per ISA 16-bit B layout.
// ---------------------------------------------------------------------------
__global__ void pack_w_bf16(const float* __restrict__ w, int N, int K,
                            int KT, int NT, uint32_t* __restrict__ out) {
  int tid = blockIdx.x * blockDim.x + threadIdx.x;
  int total = KT * NT * 32;
  if (tid >= total) return;
  int lane = tid & 31;
  int t    = tid >> 5;
  int kt   = t / NT;
  int nt   = t - kt * NT;
  int n    = nt * 16 + (lane & 15);
  int kbase = kt * 32 + ((lane < 16) ? 0 : 8);
  uint32_t* dst = out + ((size_t)t * 32 + lane) * 8;
#pragma unroll
  for (int v = 0; v < 8; ++v) {
    int k0 = kbase + 2 * (v & 3) + 16 * (v >> 2);
    int k1 = k0 + 1;
    uint16_t lo = (n < N && k0 < K) ? bf16_rne(w[(size_t)n * K + k0]) : (uint16_t)0;
    uint16_t hi = (n < N && k1 < K) ? bf16_rne(w[(size_t)n * K + k1]) : (uint16_t)0;
    dst[v] = (uint32_t)lo | ((uint32_t)hi << 16);
  }
}

// ---------------------------------------------------------------------------
// Main fused kernel: 256 threads = 8 waves, each wave owns 16 batch rows.
// LDS overlay:
//   phase A (layer0): 2 slots x [B-frags 8192B | x f32 tiles 8x2176B] = 51200B
//   phase B (layers 1..7): per-wave bf16 activation buffers 8x4224B = 33792B
// ---------------------------------------------------------------------------
#define SLOT_BYTES 25600
#define XOFF 8192

__global__ __launch_bounds__(256) void mlp_fused(
    const float* __restrict__ x, const uint32_t* __restrict__ wsb,
    const float* __restrict__ b3, const float* __restrict__ b5,
    float* __restrict__ out) {

  __shared__ __align__(16) uint8_t smem[2 * SLOT_BYTES];

  const int tid  = threadIdx.x;
  const int lane = tid & 31;
  const int wv   = tid >> 5;
  const int half = lane >> 4;   // A: K-half select; C: row-half select
  const int l15  = lane & 15;
  const int baseRow = blockIdx.x * 128 + wv * 16;

#if HAVE_TDM
  const uint32_t smem_base = (uint32_t)(size_t)&smem[0];
  auto issue = [&](int kt, int slot) {
    if (kt >= 25) return;
    uint32_t so = smem_base + (uint32_t)slot * SLOT_BYTES;
    tdm_x(so + XOFF + (uint32_t)wv * 2176,
          x + (size_t)baseRow * 784 + kt * 32, 784 - kt * 32);
    if (wv == 0)
      tdm_1d(so, wsb + OFF0 + (size_t)kt * 2048, 2048);
  };
#else
  auto issue = [&](int kt, int slot) {
    if (kt >= 25) return;
    uint8_t* so = smem + (size_t)slot * SLOT_BYTES;
    // x tile -> f32 LDS, row stride 34 dwords (coalesced float4 loads).
    float* xs = (float*)(so + XOFF) + wv * 544;
    const int c  = lane & 7;
    const int rq = lane >> 3;
    const int k0 = kt * 32 + c * 4;
#pragma unroll
    for (int p = 0; p < 4; ++p) {
      int row = p * 4 + rq;
      float4 val = make_float4(0.f, 0.f, 0.f, 0.f);
      if (k0 + 3 < 784)
        val = *(const float4*)(x + (size_t)(baseRow + row) * 784 + k0);
      float* d = xs + row * 34 + c * 4;
      *(float2*)(d + 0) = make_float2(val.x, val.y);
      *(float2*)(d + 2) = make_float2(val.z, val.w);
    }
    // B fragments (2048 dwords) cooperatively.
    const uint4* src = (const uint4*)(wsb + OFF0 + (size_t)kt * 2048 + tid * 8);
    uint4* dst = (uint4*)(so) + tid * 2;
    dst[0] = src[0];
    dst[1] = src[1];
  };
#endif

  // A fragment from padded f32 x tile (ISA 16-bit A layout), bf16-packing
  // via v_perm on the fly.
  auto loadA_x = [&](const float* xs) {
    Frag a;
#pragma unroll
    for (int v = 0; v < 8; ++v) {
      int k = ((v & 3) << 1) + ((v >> 2) << 4) + (half << 3);
      const float* p = xs + l15 * 34 + k;
      a.u[v] = pack_bf16(p[0], p[1]);
    }
    return a;
  };

  v8f acc0[8];
#pragma unroll
  for (int i = 0; i < 8; ++i) acc0[i] = (v8f)(0.0f);

  // -------------------- Layer 0: [16,784] x [784,124->128] --------------------
#if HAVE_TDM
  issue(0, 0);  // prologue: slot 0 in flight
#endif
  for (int kt = 0; kt < 25; ++kt) {
    const int slot = kt & 1;
    __syncthreads();  // all waves done reading the slot we are about to refill
#if HAVE_TDM
    issue(kt + 1, slot ^ 1);  // overlap next tile's DMA with this tile's math
    if (kt < 24) {
      if (wv == 0) __builtin_amdgcn_s_wait_tensorcnt(2);  // x+B of kt done
      else         __builtin_amdgcn_s_wait_tensorcnt(1);  // x of kt done
    } else {
      __builtin_amdgcn_s_wait_tensorcnt(0);
    }
#else
    issue(kt, slot);
#endif
    __syncthreads();  // staged data visible to all waves

    const uint8_t* so = smem + (size_t)slot * SLOT_BYTES;
    const float* xs = (const float*)(so + XOFF) + wv * 544;
    const uint32_t* bst = (const uint32_t*)so;

    Frag a = loadA_x(xs);
#pragma unroll
    for (int nt = 0; nt < 8; ++nt) {
      Frag b;
      const uint4* p = (const uint4*)&bst[(nt * 32 + lane) * 8];
      *(uint4*)&b.u[0] = p[0];
      *(uint4*)&b.u[4] = p[1];
      acc0[nt] = wmma_bf16(a, b, acc0[nt]);
    }
  }

  // -------------------- Switch LDS to activation buffers ----------------------
  __syncthreads();
  uint16_t* hb = (uint16_t*)smem + (size_t)wv * 2112;  // 16 rows x stride 132

#pragma unroll
  for (int nt = 0; nt < 8; ++nt)
#pragma unroll
    for (int r = 0; r < 8; ++r)
      hb[(half * 8 + r) * 132 + nt * 16 + l15] = bf16_1(acc0[nt][r]);
  __syncthreads();

  auto loadA_h = [&](int kt) {
    Frag a;
#pragma unroll
    for (int v = 0; v < 8; ++v) {
      int k = kt * 32 + ((v & 3) << 1) + ((v >> 2) << 4) + (half << 3);
      a.u[v] = *(const uint32_t*)&hb[l15 * 132 + k];
    }
    return a;
  };
  auto loadB_g = [&](int off, int NT, int kt, int nt) {
    Frag b;
    const uint4* p = (const uint4*)(wsb + off + (size_t)((kt * NT + nt) * 32 + lane) * 8);
    *(uint4*)&b.u[0] = p[0];
    *(uint4*)&b.u[4] = p[1];
    return b;
  };

  // -------------------- Layer 1: [16,128] x [128,32], relu --------------------
  v8f acc1[2];
#pragma unroll
  for (int i = 0; i < 2; ++i) acc1[i] = (v8f)(0.0f);
#pragma unroll
  for (int kt = 0; kt < 4; ++kt) {
    Frag a = loadA_h(kt);
#pragma unroll
    for (int nt = 0; nt < 2; ++nt)
      acc1[nt] = wmma_bf16(a, loadB_g(OFF1, 2, kt, nt), acc1[nt]);
  }
  __syncthreads();
#pragma unroll
  for (int nt = 0; nt < 2; ++nt)
#pragma unroll
    for (int r = 0; r < 8; ++r)
      hb[(half * 8 + r) * 132 + nt * 16 + l15] = bf16_1(fmaxf(acc1[nt][r], 0.f));
  __syncthreads();

  // -------------------- Layer 3: [16,32] x [32,22], +b3, relu -----------------
  {
    v8f acc[2];
#pragma unroll
    for (int i = 0; i < 2; ++i) acc[i] = (v8f)(0.0f);
    Frag a = loadA_h(0);
#pragma unroll
    for (int nt = 0; nt < 2; ++nt)
      acc[nt] = wmma_bf16(a, loadB_g(OFF3, 2, 0, nt), acc[nt]);
    __syncthreads();
#pragma unroll
    for (int nt = 0; nt < 2; ++nt) {
      int col = nt * 16 + l15;
      float bb = (col < 22) ? b3[col] : 0.f;
#pragma unroll
      for (int r = 0; r < 8; ++r) {
        float v = (col < 22) ? fmaxf(acc[nt][r] + bb, 0.f) : 0.f;
        hb[(half * 8 + r) * 132 + col] = bf16_1(v);
      }
    }
    __syncthreads();
  }

  // -------------------- Layer 5: [16,22->32] x [22,21], +b5, relu -------------
  {
    v8f acc[2];
#pragma unroll
    for (int i = 0; i < 2; ++i) acc[i] = (v8f)(0.0f);
    Frag a = loadA_h(0);
#pragma unroll
    for (int nt = 0; nt < 2; ++nt)
      acc[nt] = wmma_bf16(a, loadB_g(OFF5, 2, 0, nt), acc[nt]);
    __syncthreads();
#pragma unroll
    for (int nt = 0; nt < 2; ++nt) {
      int col = nt * 16 + l15;
      float bb = (col < 21) ? b5[col] : 0.f;
#pragma unroll
      for (int r = 0; r < 8; ++r) {
        float v = (col < 21) ? fmaxf(acc[nt][r] + bb, 0.f) : 0.f;
        hb[(half * 8 + r) * 132 + col] = bf16_1(v);
      }
    }
    __syncthreads();
  }

  // -------------------- Layer 7: [16,21->32] x [21,10] ------------------------
  v8f logits = (v8f)(0.0f);
  {
    Frag a = loadA_h(0);
    logits = wmma_bf16(a, loadB_g(OFF7, 1, 0, 0), logits);
  }

  // -------------------- log_softmax over cols 0..9 (16-lane groups) -----------
#pragma unroll
  for (int r = 0; r < 8; ++r) {
    float v   = logits[r];
    int   col = l15;
    float vm  = (col < 10) ? v : -3.402823e38f;
#pragma unroll
    for (int m = 1; m < 16; m <<= 1) vm = fmaxf(vm, __shfl_xor(vm, m, 16));
    float e = (col < 10) ? __expf(v - vm) : 0.f;
#pragma unroll
    for (int m = 1; m < 16; m <<= 1) e += __shfl_xor(e, m, 16);
    float res = v - vm - __logf(e);
    if (col < 10) {
      int row = baseRow + half * 8 + r;
      out[(size_t)row * 10 + col] = res;
    }
  }
}

// ---------------------------------------------------------------------------
extern "C" void kernel_launch(void* const* d_in, const int* in_sizes, int n_in,
                              void* d_out, int out_size, void* d_ws, size_t ws_size,
                              hipStream_t stream) {
  const float* x  = (const float*)d_in[0];
  const float* w0 = (const float*)d_in[1];
  const float* w1 = (const float*)d_in[2];
  const float* w3 = (const float*)d_in[3];
  const float* b3 = (const float*)d_in[4];
  const float* w5 = (const float*)d_in[5];
  const float* b5 = (const float*)d_in[6];
  const float* w7 = (const float*)d_in[7];
  uint32_t* wsb = (uint32_t*)d_ws;
  float* out = (float*)d_out;

  auto blocks = [](int total) { return (total + 255) / 256; };
  pack_w_bf16<<<blocks(25 * 8 * 32), 256, 0, stream>>>(w0, 124, 784, 25, 8, wsb + OFF0);
  pack_w_bf16<<<blocks(4 * 2 * 32),  256, 0, stream>>>(w1, 32, 124, 4, 2, wsb + OFF1);
  pack_w_bf16<<<blocks(1 * 2 * 32),  256, 0, stream>>>(w3, 22, 32, 1, 2, wsb + OFF3);
  pack_w_bf16<<<blocks(1 * 2 * 32),  256, 0, stream>>>(w5, 21, 22, 1, 2, wsb + OFF5);
  pack_w_bf16<<<blocks(1 * 1 * 32),  256, 0, stream>>>(w7, 10, 21, 1, 1, wsb + OFF7);

  mlp_fused<<<512, 256, 0, stream>>>(x, wsb, b3, b5, out);
}
